// CustomModel_21019569946955
// MI455X (gfx1250) — matrix operand order repaired and checked by reading, not compiled, not run on hardware
//
#include <hip/hip_runtime.h>
#include <hip/hip_bf16.h>
#include <cstdint>

// ---------------- types ----------------
typedef __attribute__((ext_vector_type(16))) __bf16 v16bf;
typedef __attribute__((ext_vector_type(8)))  __bf16 bf16x8;
typedef __attribute__((ext_vector_type(8)))  float  v8f;

// ---------------- problem constants ----------------
#define HDIM 2048
#define BT   32            // batch rows per workgroup
#define NTPW 16            // 16x16 N-tiles per wave (8 waves * 256 cols = 2048)

// pendulum / loss constants
#define Gc   9.8f
#define C1c  0.3f          // L*MB
#define C2c  2.09f         // I + L*L*MB
#define C3c  4.0f          // MB+MC
#define ATc  0.2f
#define ARc  0.2f
#define PENc 10000.0f
#define ALPHAc 0.1f
#define EPSc 1e-07f

// LDS layout (bytes)
#define SM_H1   0                         // [32][2048] bf16 = 131072
#define SM_G2   131072                    // [32][2048] bf16 = 131072
#define SM_YPP  262144                    // float [8][32]
#define SM_VVP  (SM_YPP + 8*32*4)         // float [8][32]
#define SM_DVP  (SM_VVP + 8*32*4)         // float [8][32][4]
#define SM_BYTES (SM_DVP + 8*32*4*4)      // = 268288

// ---------------- fast tanh: single V_TANH_F32 on gfx1250 ----------------
__device__ __forceinline__ float fast_tanh(float v) {
#if __has_builtin(__builtin_amdgcn_tanhf)
    return __builtin_amdgcn_tanhf(v);
#else
    return tanhf(v);
#endif
}

// ---------------- WMMA fragment loaders ----------------
// A (16x32 bf16), row-major source with row stride HDIM.
// ISA layout: lane = M (l&15), half h=l>>4; halfs v<8 -> K=8h+v ; v>=8 -> K=16+8h+(v-8)
__device__ __forceinline__ v16bf ldA(const __bf16* tile, int lane) {
    const __bf16* p = tile + (lane & 15) * HDIM + ((lane >> 4) << 3);
    bf16x8 lo = *reinterpret_cast<const bf16x8*>(p);
    bf16x8 hi = *reinterpret_cast<const bf16x8*>(p + 16);
    v16bf a;
#pragma unroll
    for (int i = 0; i < 8; ++i) { a[i] = lo[i]; a[i + 8] = hi[i]; }
    return a;
}

// B (32x16 bf16): B[K, col] = mat[col0+col][k0+K], mat row-major stride HDIM.
// ISA layout: lane = col (l&15); halfs v<8 -> K=8h+v ; v>=8 -> K=16+8h+(v-8)
__device__ __forceinline__ v16bf ldB(const __bf16* mat, int col0, int k0, int lane) {
    const __bf16* p = mat + (size_t)(col0 + (lane & 15)) * HDIM + k0 + ((lane >> 4) << 3);
    bf16x8 lo = *reinterpret_cast<const bf16x8*>(p);
    bf16x8 hi = *reinterpret_cast<const bf16x8*>(p + 16);
    v16bf b;
#pragma unroll
    for (int i = 0; i < 8; ++i) { b[i] = lo[i]; b[i + 8] = hi[i]; }
    return b;
}

// ---------------- kernel 0: W2 f32 -> bf16 (row-major) + bf16 transpose ----------------
__global__ __launch_bounds__(256) void k_convert_w2(const float* __restrict__ W2,
                                                    __bf16* __restrict__ w2bf,
                                                    __bf16* __restrict__ w2t) {
    int i = blockIdx.x * 256 + threadIdx.x;      // grid covers exactly H*H
    float w = W2[i];
    int r = i >> 11, c = i & (HDIM - 1);
    __bf16 v = (__bf16)w;
    w2bf[i] = v;
    w2t[(size_t)c * HDIM + r] = v;
}

// ---------------- kernel 1: fused fwd + bwd + per-row terms ----------------
__global__ __launch_bounds__(256) void k_fused(
    const float* __restrict__ x, const float* __restrict__ y,
    const float* __restrict__ W1, const float* __restrict__ b1,
    const float* __restrict__ b2, const float* __restrict__ W3,
    const float* __restrict__ b3,
    const __bf16* __restrict__ w2bf,   // W2 row-major bf16  (backward B operand)
    const __bf16* __restrict__ w2t,    // W2^T row-major bf16 (forward B operand)
    float* __restrict__ pen, float* __restrict__ partA, float* __restrict__ partB) {

    extern __shared__ char smem[];
    __bf16* h1s = (__bf16*)(smem + SM_H1);
    __bf16* g2s = (__bf16*)(smem + SM_G2);
    float*  ypP = (float*)(smem + SM_YPP);
    float*  vvP = (float*)(smem + SM_VVP);
    float*  dvP = (float*)(smem + SM_DVP);

    const int tid  = threadIdx.x;
    const int lane = tid & 31;
    const int wv   = tid >> 5;
    const int hh   = lane >> 4;
    const int row0 = blockIdx.x * BT;
    const int ncol0 = wv * (NTPW * 16);

    // ---- Phase 0: h1 = tanh(x @ W1 + b1) into LDS (bf16) ----
    for (int e = tid; e < BT * HDIM; e += 256) {
        int r = e >> 11, c = e & (HDIM - 1);
        const float* xr = x + (size_t)(row0 + r) * 4;
        float z = b1[c] + xr[0] * W1[c] + xr[1] * W1[HDIM + c]
                        + xr[2] * W1[2 * HDIM + c] + xr[3] * W1[3 * HDIM + c];
        h1s[r * HDIM + c] = (__bf16)fast_tanh(z);
    }
    __syncthreads();

    // ---- Phase 1: h2 tiles = tanh(h1 @ W2 + b2); emit g2 to LDS, y_pred/V partials ----
    float yp[2][8], vv[2][8];
#pragma unroll
    for (int mt = 0; mt < 2; ++mt)
#pragma unroll
        for (int i = 0; i < 8; ++i) { yp[mt][i] = 0.f; vv[mt][i] = 0.f; }

    for (int t = 0; t < NTPW; ++t) {
        int n0 = ncol0 + t * 16;
        v8f accs[2] = {v8f{}, v8f{}};
#pragma unroll 2
        for (int k0 = 0; k0 < HDIM; k0 += 32) {
            v16bf bm = ldB(w2t, n0, k0, lane);
            v16bf a0 = ldA(h1s + k0, lane);
            v16bf a1 = ldA(h1s + 16 * HDIM + k0, lane);
            accs[0] = __builtin_amdgcn_wmma_f32_16x16x32_bf16(false, a0, false, bm,
                                                              (short)0, accs[0], false, false);
            accs[1] = __builtin_amdgcn_wmma_f32_16x16x32_bf16(false, a1, false, bm,
                                                              (short)0, accs[1], false, false);
        }
        int n = n0 + (lane & 15);
        float w3a = W3[2 * n], w3b = W3[2 * n + 1], bb = b2[n];
#pragma unroll
        for (int mt = 0; mt < 2; ++mt)
#pragma unroll
            for (int i = 0; i < 8; ++i) {
                float h2 = fast_tanh(accs[mt][i] + bb);
                yp[mt][i] += h2 * w3a;
                vv[mt][i] += h2 * w3b;
                int r = mt * 16 + i + 8 * hh;
                g2s[r * HDIM + n] = (__bf16)(w3b * (1.0f - h2 * h2));
            }
    }
    // reduce y_pred / V partials across the 16-lane half (lanes share rows within a half)
#pragma unroll
    for (int mt = 0; mt < 2; ++mt)
#pragma unroll
        for (int i = 0; i < 8; ++i) {
            float a = yp[mt][i], v = vv[mt][i];
#pragma unroll
            for (int m = 8; m >= 1; m >>= 1) {
                a += __shfl_xor(a, m, 16);
                v += __shfl_xor(v, m, 16);
            }
            if ((lane & 15) == 0) {
                int r = mt * 16 + i + 8 * hh;
                ypP[wv * 32 + r] = a;
                vvP[wv * 32 + r] = v;
            }
        }
    __syncthreads();

    // ---- Phase 2: g1 = g2 @ W2^T ; fold (1-h1^2) and W1^T -> dVdx partials ----
    float dv[2][8][4];
#pragma unroll
    for (int mt = 0; mt < 2; ++mt)
#pragma unroll
        for (int i = 0; i < 8; ++i)
#pragma unroll
            for (int k = 0; k < 4; ++k) dv[mt][i][k] = 0.f;

    for (int t = 0; t < NTPW; ++t) {
        int j0 = ncol0 + t * 16;
        v8f accs[2] = {v8f{}, v8f{}};
#pragma unroll 2
        for (int n0 = 0; n0 < HDIM; n0 += 32) {
            v16bf bm = ldB(w2bf, j0, n0, lane);
            v16bf a0 = ldA(g2s + n0, lane);
            v16bf a1 = ldA(g2s + 16 * HDIM + n0, lane);
            accs[0] = __builtin_amdgcn_wmma_f32_16x16x32_bf16(false, a0, false, bm,
                                                              (short)0, accs[0], false, false);
            accs[1] = __builtin_amdgcn_wmma_f32_16x16x32_bf16(false, a1, false, bm,
                                                              (short)0, accs[1], false, false);
        }
        int j = j0 + (lane & 15);
        float w10 = W1[j], w11 = W1[HDIM + j], w12 = W1[2 * HDIM + j], w13 = W1[3 * HDIM + j];
#pragma unroll
        for (int mt = 0; mt < 2; ++mt)
#pragma unroll
            for (int i = 0; i < 8; ++i) {
                int r = mt * 16 + i + 8 * hh;
                float h1v = (float)h1s[r * HDIM + j];
                float dz = accs[mt][i] * (1.0f - h1v * h1v);
                dv[mt][i][0] += dz * w10;
                dv[mt][i][1] += dz * w11;
                dv[mt][i][2] += dz * w12;
                dv[mt][i][3] += dz * w13;
            }
    }
#pragma unroll
    for (int mt = 0; mt < 2; ++mt)
#pragma unroll
        for (int i = 0; i < 8; ++i)
#pragma unroll
            for (int k = 0; k < 4; ++k) {
                float a = dv[mt][i][k];
#pragma unroll
                for (int m = 8; m >= 1; m >>= 1) a += __shfl_xor(a, m, 16);
                if ((lane & 15) == 0) {
                    int r = mt * 16 + i + 8 * hh;
                    dvP[wv * 128 + r * 4 + k] = a;
                }
            }
    __syncthreads();

    // ---- Phase 3: per-row pendulum / penalty + block loss partials (wave 0 only) ----
    if (tid < 32) {
        int r = tid;
        float ypred = b3[0], V = b3[1];
        float dvdx[4] = {0.f, 0.f, 0.f, 0.f};
#pragma unroll
        for (int w = 0; w < 8; ++w) {
            ypred += ypP[w * 32 + r];
            V     += vvP[w * 32 + r];
#pragma unroll
            for (int k = 0; k < 4; ++k) dvdx[k] += dvP[w * 128 + r * 4 + k];
        }
        const float* xr = x + (size_t)(row0 + r) * 4;
        float x2 = xr[1], x3 = xr[2], x4 = xr[3];
        float cc = cosf(x3), ss = sinf(x3);
        float den = C2c * C3c - C1c * C1c * cc * cc;
        float f = ypred - ATc * x2;
        float x2p = (Gc * C1c * C1c * cc * ss + C2c * f - ARc * C1c * cc * x4
                     - C1c * C2c * ss * x4 * x4) / den;
        float x4p = (Gc * C1c * C3c * ss + C1c * cc * f - ARc * C3c * x4
                     - C1c * C1c * cc * ss * x4 * x4) / den;
        float Vdot = dvdx[0] * x2 + dvdx[1] * x2p + dvdx[2] * x4 + dvdx[3] * x4p;
        pen[row0 + r] = PENc * fmaxf(-V, 0.f) + PENc * fmaxf(Vdot, 0.f);

        float yv = y[row0 + r];
        float d = log1pf(fmaxf(ypred, EPSc)) - log1pf(fmaxf(yv, EPSc));
        float sa = d * d;
        float sb = (yv - ypred) * (yv - ypred);
#pragma unroll
        for (int m = 16; m >= 1; m >>= 1) {
            sa += __shfl_xor(sa, m, 32);
            sb += __shfl_xor(sb, m, 32);
        }
        if (tid == 0) { partA[blockIdx.x] = sa; partB[blockIdx.x] = sb; }
    }
}

// ---------------- kernel 2: reduce block partials -> scalar loss term ----------------
__global__ __launch_bounds__(256) void k_reduce(const float* __restrict__ partA,
                                                const float* __restrict__ partB,
                                                float* __restrict__ scalar,
                                                int n, float invB) {
    __shared__ float sA[256], sB[256];
    float a = 0.f, b = 0.f;
    for (int i = threadIdx.x; i < n; i += 256) { a += partA[i]; b += partB[i]; }
    sA[threadIdx.x] = a; sB[threadIdx.x] = b;
    __syncthreads();
    for (int s = 128; s > 0; s >>= 1) {
        if (threadIdx.x < (unsigned)s) {
            sA[threadIdx.x] += sA[threadIdx.x + s];
            sB[threadIdx.x] += sB[threadIdx.x + s];
        }
        __syncthreads();
    }
    if (threadIdx.x == 0)
        scalar[0] = ALPHAc * (sA[0] * invB) + (1.0f - ALPHAc) * (sB[0] * invB);
}

// ---------------- kernel 3: out[i] = scalar + pen[i] ----------------
__global__ __launch_bounds__(256) void k_finalize(const float* __restrict__ pen,
                                                  const float* __restrict__ scalar,
                                                  float* __restrict__ out) {
    int i = blockIdx.x * 256 + threadIdx.x;
    out[i] = scalar[0] + pen[i];
}

// ---------------- host glue ----------------
extern "C" void kernel_launch(void* const* d_in, const int* in_sizes, int n_in,
                              void* d_out, int out_size, void* d_ws, size_t ws_size,
                              hipStream_t stream) {
    (void)n_in; (void)out_size; (void)ws_size;
    const float* x  = (const float*)d_in[0];
    const float* y  = (const float*)d_in[1];
    const float* W1 = (const float*)d_in[2];
    const float* b1 = (const float*)d_in[3];
    const float* W2 = (const float*)d_in[4];
    const float* b2 = (const float*)d_in[5];
    const float* W3 = (const float*)d_in[6];
    const float* b3 = (const float*)d_in[7];
    float* out = (float*)d_out;

    const int B = in_sizes[1];               // 65536
    const size_t HH = (size_t)HDIM * HDIM;   // 4M

    // workspace layout
    __bf16* w2bf = (__bf16*)d_ws;            // 8 MB
    __bf16* w2t  = w2bf + HH;                // 8 MB
    float*  pen  = (float*)(w2t + HH);       // B floats
    float*  partA = pen + B;                 // B/32
    float*  partB = partA + (B / BT);
    float*  scalar = partB + (B / BT);

    // allow >64KB dynamic LDS for the fused kernel
    (void)hipFuncSetAttribute((const void*)k_fused,
                              hipFuncAttributeMaxDynamicSharedMemorySize, SM_BYTES);

    k_convert_w2<<<(int)(HH / 256), 256, 0, stream>>>(W2, w2bf, w2t);
    k_fused<<<B / BT, 256, SM_BYTES, stream>>>(x, y, W1, b1, b2, W3, b3,
                                               w2bf, w2t, pen, partA, partB);
    k_reduce<<<1, 256, 0, stream>>>(partA, partB, scalar, B / BT, 1.0f / (float)B);
    k_finalize<<<B / 256, 256, 0, stream>>>(pen, scalar, out);
}
